// SJModel_88270167867785
// MI455X (gfx1250) — matrix operand order repaired and checked by reading, not compiled, not run on hardware
//
#include <hip/hip_runtime.h>
#include <cstdint>

// ---------------------------------------------------------------------------
// SNN conv3x3 (64->64) + multi-step LIF, fused.  MI455X / gfx1250, wave32.
//   T=8  B=16  C=64  H=W=64   K = 64*9 = 576
// GEMM view per image: D[64co x 4096px] = Wmat[64 x 576] * Patches[576 x 4096]
// Wave tile: 16co x 64px; all 18 A-fragments (weights) live in registers.
// ---------------------------------------------------------------------------

typedef __attribute__((ext_vector_type(16))) _Float16 v16h;
typedef __attribute__((ext_vector_type(8)))  float    v8f;
typedef __attribute__((ext_vector_type(4)))  int      v4i;

#define T_STEPS 8
#define BATCH   16
#define CH      64
#define HH      64
#define WW      64

// workspace layout
#define XH_ELEMS ((size_t)T_STEPS * BATCH * HH * WW * CH)      // NHWC f16 x
#define WS_W_OFF (XH_ELEMS)                                    // swizzled W f16

// ---- async copy to LDS (guarded: falls back to load+ds_store) -------------
#if defined(__has_builtin)
#  if __has_builtin(__builtin_amdgcn_global_load_async_to_lds_b128)
#    define HAVE_ASYNC_CP 1
#  endif
#  if __has_builtin(__builtin_amdgcn_s_wait_asynccnt)
#    define HAVE_ASYNC_WAIT 1
#  endif
#endif
#ifndef HAVE_ASYNC_CP
#  define HAVE_ASYNC_CP 0
#endif
#ifndef HAVE_ASYNC_WAIT
#  define HAVE_ASYNC_WAIT 0
#endif

#if HAVE_ASYNC_CP
typedef __attribute__((address_space(1))) v4i GlobalV4i;  // global int4
typedef __attribute__((address_space(3))) v4i LocalV4i;   // LDS int4
#endif

__device__ __forceinline__ void cp16_g2l(const _Float16* g, _Float16* l) {
#if HAVE_ASYNC_CP
  __builtin_amdgcn_global_load_async_to_lds_b128(
      (GlobalV4i*)(uintptr_t)g,
      (LocalV4i*)(unsigned int)(uintptr_t)l,
      0, 0);
#else
  *(int4*)l = *(const int4*)g;
#endif
}

__device__ __forceinline__ void async_wait0() {
#if HAVE_ASYNC_CP
#  if HAVE_ASYNC_WAIT
  __builtin_amdgcn_s_wait_asynccnt(0);
#  else
  asm volatile("s_wait_asynccnt 0" ::: "memory");
#  endif
#endif
}

// ---------------------------------------------------------------------------
// Kernel 1: x NCHW fp32  ->  NHWC f16  (LDS-tiled transpose over ci<->w)
// grid = T*B*H blocks, 256 threads
// ---------------------------------------------------------------------------
__global__ __launch_bounds__(256) void xpose_kernel(const float* __restrict__ x,
                                                    _Float16* __restrict__ xh) {
  __shared__ __align__(16) _Float16 tile[64 * 66];
  const int tb  = blockIdx.x >> 6;   // 0..127  (t*B+b)
  const int h   = blockIdx.x & 63;
  const int tid = threadIdx.x;

  const int w   = tid & 63;
  const int cib = tid >> 6;  // 0..3
  const float* src = x + (size_t)tb * (CH * HH * WW) + (size_t)h * WW;
#pragma unroll
  for (int j = 0; j < 16; ++j) {
    const int ci = cib + j * 4;
    tile[w * 66 + ci] = (_Float16)src[(size_t)ci * (HH * WW) + w];
  }
  __syncthreads();

  const int ci2 = tid & 31;
  const int w0  = tid >> 5;
  _Float16* dst = xh + ((size_t)tb * HH + h) * (WW * CH);
#pragma unroll
  for (int j = 0; j < 8; ++j) {
    const int ww = w0 + j * 8;
    const unsigned int pk = *(const unsigned int*)&tile[ww * 66 + ci2 * 2];
    *(unsigned int*)&dst[(size_t)ww * CH + ci2 * 2] = pk;
  }
}

// ---------------------------------------------------------------------------
// Kernel 2: swizzle W [64co][64ci][3][3] fp32 into per-lane WMMA A-fragment
// order, f16.  kc in 0..17: tap = kc/2 (kh*3+kw), c0 = (kc&1)*32.
// A k order used here (contiguous, matches B reads):  lane<16: K {0..15}; lane>=16: K {16..31}
// ---------------------------------------------------------------------------
__global__ __launch_bounds__(256) void wswz_kernel(const float* __restrict__ Wt,
                                                   _Float16* __restrict__ wsw) {
  const int tid = blockIdx.x * 256 + threadIdx.x;
  if (tid >= 4 * 18 * 32) return;
  const int lane = tid & 31;
  const int pos  = tid >> 5;        // cb*18 + kc
  const int kc   = pos % 18;
  const int cb   = pos / 18;
  const int tap  = kc >> 1;
  const int c0   = (kc & 1) << 5;
  const int m    = cb * 16 + (lane & 15);
  const int l16  = lane >> 4;
  v16h v;
#pragma unroll
  for (int hh = 0; hh < 16; ++hh) {
    const int k  = hh + 16 * l16;
    const int ci = c0 + k;
    v[hh] = (_Float16)Wt[(size_t)m * 576 + ci * 9 + tap];
  }
  *(v16h*)&wsw[(size_t)tid * 16] = v;
}

// ---------------------------------------------------------------------------
// Kernel 3: fused implicit-GEMM conv + 8-step LIF.
// grid = B * (H/2) = 512 blocks, 256 threads (8 waves).
// Block: fixed b, 2 output rows h0..h0+1, all 64 co.
// Wave: cb = wid&3 (co block of 16), row = wid>>2 (0..1); tile 16co x 64px.
// All 18 weight A-fragments resident in VGPRs; B-fragments double-buffered.
// LDS: Xs[4 rows][66 cols][64 ci] f16 (halo rows + zeroed halo columns).
// ---------------------------------------------------------------------------

// LDS byte-free (half-index) offset of k-chunk kc relative to ldsb
#define XOFF(kc) (((((kc) >> 1) / 3) * 66 + (((kc) >> 1) % 3)) * CH + (((kc) & 1) << 5))

__global__ __launch_bounds__(256) void conv_lif_kernel(const _Float16* __restrict__ xh,
                                                       const _Float16* __restrict__ wsw,
                                                       float* __restrict__ out) {
  __shared__ __align__(32) _Float16 Xs[4 * 66 * CH];  // 33792 B

  const int b   = blockIdx.x >> 5;          // 0..15
  const int h0  = (blockIdx.x & 31) << 1;   // 0,2,...,62
  const int tid = threadIdx.x;
  const int wid  = tid >> 5;
  const int lane = tid & 31;
  const int cb   = wid & 3;                 // co block (16 channels)
  const int row  = wid >> 2;                // 0..1 output row within stripe
  const int n    = lane & 15;               // B-matrix column / px within 16
  const int khf  = lane >> 4;               // B-matrix K half

  // zero the halo columns (c = 0 and c = 65) once; persists across timesteps
  if (tid < 64) {
    const int seg = tid >> 3;               // 0..7 = row*2 + side
    const int r   = seg >> 1;
    const int c   = (seg & 1) ? 65 : 0;
    int4 z; z.x = 0; z.y = 0; z.z = 0; z.w = 0;
    *(int4*)&Xs[(r * 66 + c) * CH + (tid & 7) * 8] = z;
  }

  // ---- resident weights: 18 A-fragments = 144 VGPRs -----------------------
  v16h af[18];
#pragma unroll
  for (int kc = 0; kc < 18; ++kc)
    af[kc] = *(const v16h*)&wsw[((size_t)(cb * 18 + kc) * 32 + lane) * 16];

  // LDS base for this lane: (row*66 + n)*64 + khf*16 halves
  const int ldsb = (row * 66 + n) * CH + khf * 16;

  v8f vstate[4] = {};   // membrane potential (WMMA C layout), persists over T

#pragma unroll 1
  for (int t = 0; t < T_STEPS; ++t) {
    // ---- stage 4 halo rows of NHWC f16 into LDS (async DMA path) ----------
#pragma unroll
    for (int j = 0; j < 8; ++j) {
      const int q   = tid + j * 256;   // 0..2047 (4 rows * 512 x 16B chunks)
      const int r   = q >> 9;
      const int c16 = q & 511;
      const int h   = h0 - 1 + r;
      _Float16* ldst = &Xs[r * (66 * CH) + CH /*col 1*/ + c16 * 8];
      if ((unsigned)h < (unsigned)HH) {
        const _Float16* g =
            xh + (((size_t)(t * BATCH + b) * HH + h) * WW) * CH + (size_t)c16 * 8;
        cp16_g2l(g, ldst);
      } else {
        int4 z; z.x = 0; z.y = 0; z.z = 0; z.w = 0;
        *(int4*)ldst = z;
      }
    }
    async_wait0();
    __syncthreads();

    // ---- implicit GEMM: K = 576 in 18 chunks of 32, software-pipelined ----
    v8f acc[4] = {};
    v16h bf[4];
#pragma unroll
    for (int p = 0; p < 4; ++p)
      bf[p] = *(const v16h*)&Xs[ldsb + XOFF(0) + p * (16 * CH)];

#pragma unroll
    for (int kc = 0; kc < 18; ++kc) {
      v16h bfn[4];
      if (kc < 17) {
#pragma unroll
        for (int p = 0; p < 4; ++p)
          bfn[p] = *(const v16h*)&Xs[ldsb + XOFF(kc + 1) + p * (16 * CH)];
      }
#pragma unroll
      for (int p = 0; p < 4; ++p)
        acc[p] = __builtin_amdgcn_wmma_f32_16x16x32_f16(
            false, af[kc], false, bf[p], (short)0, acc[p], false, false);
      if (kc < 17) {
#pragma unroll
        for (int p = 0; p < 4; ++p) bf[p] = bfn[p];
      }
    }

    // ---- LIF: v = (v + z)/2 ; s = H(v - 1) ; hard reset ; store spikes ----
    const int hg = h0 + row;
#pragma unroll
    for (int p = 0; p < 4; ++p) {
      v8f z = acc[p];
      v8f v = vstate[p];
      const int col = p * 16 + n;
      float* outp = out + ((((size_t)t * BATCH + b) * CH + (cb * 16 + khf * 8)) * HH + hg) * WW + col;
#pragma unroll
      for (int r8 = 0; r8 < 8; ++r8) {
        const float vv = 0.5f * (v[r8] + z[r8]);
        const bool fire = (vv >= 1.0f);
        outp[(size_t)r8 * (HH * WW)] = fire ? 1.0f : 0.0f;
        v[r8] = fire ? 0.0f : vv;
      }
      vstate[p] = v;
    }
    __syncthreads();   // protect Xs before next timestep's staging
  }
}

// ---------------------------------------------------------------------------
extern "C" void kernel_launch(void* const* d_in, const int* in_sizes, int n_in,
                              void* d_out, int out_size, void* d_ws, size_t ws_size,
                              hipStream_t stream) {
  (void)in_sizes; (void)n_in; (void)out_size; (void)ws_size;
  const float* x  = (const float*)d_in[0];   // [8][16][64][64][64] f32
  const float* Wt = (const float*)d_in[1];   // [64][64][3][3] f32
  float* out = (float*)d_out;                // [8][16][64][64][64] f32 spikes

  _Float16* xh  = (_Float16*)d_ws;                       // 64 MiB NHWC f16
  _Float16* wsw = (_Float16*)d_ws + WS_W_OFF;            // 72 KiB swizzled W

  xpose_kernel<<<T_STEPS * BATCH * HH, 256, 0, stream>>>(x, xh);
  wswz_kernel<<<9, 256, 0, stream>>>(Wt, wsw);
  conv_lif_kernel<<<BATCH * (HH / 2), 256, 0, stream>>>(xh, wsw, out);
}
